// DynamicKernelFusion_69303592288826
// MI455X (gfx1250) — compile-verified, matrix operand order checked by reading, not run on hardware
//
#include <hip/hip_runtime.h>

// ---- problem sizes (fixed by the reference) ----
#define D_DIM  1024
#define KFOLD  16
#define M_TOT  8192   // B*L = 4*2048
#define K_TOT  3072   // 3*D
#define N_TOT  1024   // D

// ---- GEMM tiling ----
#define BM 128
#define BN 128
#define BK 32
#define NKT (K_TOT / BK)   // 96
#define LDA 40   // padded LDS row pitch in halfs (80 B; 16B-aligned chunks, conflict-free)
#define LDB 40
#define NT 256   // 8 waves per block (wave32); wave grid 4(M) x 2(N), wave tile 32x64

typedef _Float16 v16h __attribute__((ext_vector_type(16)));
typedef _Float16 v8h  __attribute__((ext_vector_type(8)));
typedef _Float16 h4   __attribute__((ext_vector_type(4)));
typedef float    v8f  __attribute__((ext_vector_type(8)));
typedef int      gv4i __attribute__((vector_size(16)));   // matches builtin's V4i

#if defined(__gfx1250__) && __has_builtin(__builtin_amdgcn_global_load_async_to_lds_b128)
#define HAVE_ASYNC 1
#else
#define HAVE_ASYNC 0
#endif

union Frag { v16h v; v8h h[2]; };

__device__ __forceinline__ void wait_async_lds() {
#if HAVE_ASYNC
#if __has_builtin(__builtin_amdgcn_s_wait_asynccnt)
  __builtin_amdgcn_s_wait_asynccnt(0);
#else
  asm volatile("s_wait_asynccnt 0x0" ::: "memory");
#endif
#endif
}

// ---------- fold W: W'(d,c) = sum_{k<16} W[d*16+k, c], stored f16 row-major [N][K] ----------
__global__ __launch_bounds__(NT) void fold_w_kernel(const float* __restrict__ W,
                                                    _Float16* __restrict__ Wf) {
  const int idx = blockIdx.x * NT + threadIdx.x;   // 0 .. N_TOT*K_TOT-1
  const int c = idx % K_TOT;
  const int d = idx / K_TOT;
  const float* p = W + (size_t)d * KFOLD * K_TOT + c;
  float s = 0.f;
#pragma unroll
  for (int k = 0; k < KFOLD; ++k) s += p[(size_t)k * K_TOT];
  Wf[(size_t)d * K_TOT + c] = (_Float16)s;
}

__global__ __launch_bounds__(NT) void fold_b_kernel(const float* __restrict__ b,
                                                    float* __restrict__ bsum) {
  const int d = blockIdx.x * NT + threadIdx.x;
  float s = 0.f;
#pragma unroll
  for (int k = 0; k < KFOLD; ++k) s += b[d * KFOLD + k];
  bsum[d] = s;
}

// ---------- fused GEMM: out[m,n] = (A(m,:)·W'(n,:) + b'(n)) * x[m,n] * y[m,n] ----------
__global__ __launch_bounds__(NT) void gemm_fused_kernel(
    const float* __restrict__ x, const float* __restrict__ y,
    const float* __restrict__ z, const _Float16* __restrict__ Wf,
    const float* __restrict__ bsum, float* __restrict__ out) {
  __shared__ __align__(16) _Float16 Als[2][BM * LDA];
  __shared__ __align__(16) _Float16 Bls[2][BN * LDB];

  const int tid  = threadIdx.x;
  const int m0   = blockIdx.y * BM;
  const int n0   = blockIdx.x * BN;
  const int wid  = tid >> 5;          // 8 waves
  const int lane = tid & 31;
  const int lrow = lane & 15;
  const int lhi  = lane >> 4;
  const int wm   = (wid >> 1) << 5;   // wave M offset: 0/32/64/96
  const int wn   = (wid & 1) << 6;    // wave N offset: 0/64

  v8f acc[2][4] = {};
  float4 va[4];                        // A prefetch registers (one 128x32 f32 tile / block)
  Frag af[2], bfr[4];

  // ---- staging helpers ----
  auto issueA = [&](int k0) {          // global f32 loads -> va[]
    const float* srcA = (k0 < D_DIM) ? x : (k0 < 2 * D_DIM) ? y : z;
    const int koff = k0 & (D_DIM - 1);
#pragma unroll
    for (int i = 0; i < 4; ++i) {
      const int linear = i * NT + tid;        // float4-chunk id, 0..1023
      const int row = linear >> 3;
      const int c4  = linear & 7;
      va[i] = *reinterpret_cast<const float4*>(
          srcA + (size_t)(m0 + row) * D_DIM + koff + c4 * 4);
    }
  };
  auto commitA = [&](int buf) {        // f32 -> f16 -> LDS
#pragma unroll
    for (int i = 0; i < 4; ++i) {
      const int linear = i * NT + tid;
      const int row = linear >> 3;
      const int c4  = linear & 7;
      h4 hv;
      hv.x = (_Float16)va[i].x; hv.y = (_Float16)va[i].y;
      hv.z = (_Float16)va[i].z; hv.w = (_Float16)va[i].w;
      *reinterpret_cast<h4*>(&Als[buf][row * LDA + c4 * 4]) = hv;
    }
  };
  auto issueB = [&](int k0, int buf) { // f16 workspace -> LDS (async direct-to-LDS)
#pragma unroll
    for (int i = 0; i < 2; ++i) {
      const int c   = i * NT + tid;           // 16B-chunk id, 0..511
      const int row = c >> 2;
      const int seg = c & 3;
      const _Float16* gB = Wf + (size_t)(n0 + row) * K_TOT + k0 + seg * 8;
      _Float16* lB = &Bls[buf][row * LDB + seg * 8];
#if HAVE_ASYNC
      __builtin_amdgcn_global_load_async_to_lds_b128(
          (__attribute__((address_space(1))) gv4i*)gB,
          (__attribute__((address_space(3))) gv4i*)lB, 0, 0);
#else
      *reinterpret_cast<v8h*>(lB) = *reinterpret_cast<const v8h*>(gB);
#endif
    }
  };
  auto loadFrags = [&](int buf) {      // ISA 16-bit layout: runs [8h,8h+8) and [16+8h,+8)
#pragma unroll
    for (int im = 0; im < 2; ++im) {
      const int r = wm + im * 16 + lrow;
      af[im].h[0] = *reinterpret_cast<const v8h*>(&Als[buf][r * LDA + lhi * 8]);
      af[im].h[1] = *reinterpret_cast<const v8h*>(&Als[buf][r * LDA + 16 + lhi * 8]);
    }
#pragma unroll
    for (int in = 0; in < 4; ++in) {
      const int r = wn + in * 16 + lrow;
      bfr[in].h[0] = *reinterpret_cast<const v8h*>(&Bls[buf][r * LDB + lhi * 8]);
      bfr[in].h[1] = *reinterpret_cast<const v8h*>(&Bls[buf][r * LDB + 16 + lhi * 8]);
    }
  };

  // ---- software-pipelined, double-buffered main loop ----
  issueB(0, 0);
  issueA(0);
  commitA(0);
  wait_async_lds();
  __syncthreads();
  loadFrags(0);

  for (int kt = 0; kt < NKT; ++kt) {
    const int nxt  = (kt + 1) & 1;
    const bool more = (kt + 1) < NKT;
    if (more) {                       // prefetch next tile before the matrix math
      issueB((kt + 1) * BK, nxt);
      issueA((kt + 1) * BK);
    }
#pragma unroll
    for (int im = 0; im < 2; ++im)
#pragma unroll
      for (int in = 0; in < 4; ++in)
        acc[im][in] = __builtin_amdgcn_wmma_f32_16x16x32_f16(
            false, af[im].v, false, bfr[in].v, (short)0, acc[im][in], false, false);
    if (more) {
      commitA(nxt);
      wait_async_lds();
      __syncthreads();
      loadFrags(nxt);
    }
  }

  // ---- fused epilogue: (acc + bias) * x * y ; C/D layout: M=r+8*lhi, N=lrow ----
#pragma unroll
  for (int in = 0; in < 4; ++in) {
    const int n = n0 + wn + in * 16 + lrow;
    const float bias = bsum[n];
#pragma unroll
    for (int im = 0; im < 2; ++im) {
#pragma unroll
      for (int r = 0; r < 8; ++r) {
        const int m = m0 + wm + im * 16 + lhi * 8 + r;
        const size_t o = (size_t)m * N_TOT + n;
        out[o] = (acc[im][in][r] + bias) * x[o] * y[o];
      }
    }
  }
}

extern "C" void kernel_launch(void* const* d_in, const int* in_sizes, int n_in,
                              void* d_out, int out_size, void* d_ws, size_t ws_size,
                              hipStream_t stream) {
  const float* x = (const float*)d_in[0];
  const float* y = (const float*)d_in[1];
  const float* z = (const float*)d_in[2];
  const float* W = (const float*)d_in[3];
  const float* b = (const float*)d_in[4];
  float* out = (float*)d_out;

  _Float16* Wf = (_Float16*)d_ws;                                   // 6.29 MB
  float* bsum = (float*)((char*)d_ws + (size_t)N_TOT * K_TOT * sizeof(_Float16));

  fold_w_kernel<<<(N_TOT * K_TOT) / NT, NT, 0, stream>>>(W, Wf);
  fold_b_kernel<<<N_TOT / NT, NT, 0, stream>>>(b, bsum);

  dim3 grid(N_TOT / BN, M_TOT / BM);   // 8 x 64 blocks
  gemm_fused_kernel<<<grid, NT, 0, stream>>>(x, y, z, Wf, bsum, out);
}
